// OriginalTemporalSAE_73907797229709
// MI455X (gfx1250) — matrix operand order repaired and checked by reading, not compiled, not run on hardware
//
#include <hip/hip_runtime.h>
#include <cstdint>

// ---------------------------------------------------------------------------
// TemporalSAE forward for MI455X (gfx1250, wave32, WMMA + TDM async tiles).
// ---------------------------------------------------------------------------

typedef __bf16 bf16;
typedef unsigned int       u32;
typedef unsigned long long u64;
typedef __attribute__((ext_vector_type(8)))  bf16  v8bf;
typedef __attribute__((ext_vector_type(16))) bf16  v16bf;
typedef __attribute__((ext_vector_type(8)))  float v8f;
typedef __attribute__((ext_vector_type(4)))  u32   v4u;
typedef __attribute__((ext_vector_type(8)))  int   v8i;
typedef __attribute__((ext_vector_type(4)))  int   v4i;

#define BATCH   4
#define SEQL    2048
#define DIMIN   512
#define WIDTH   4096
#define NHEADS  8
#define NE      64
#define DH      8
#define DV      512
#define MROWS   (BATCH * SEQL)   // 8192

#if defined(__has_builtin)
#if __has_builtin(__builtin_amdgcn_tensor_load_to_lds) && __has_builtin(__builtin_amdgcn_s_wait_tensorcnt)
#define HAVE_TDM 1
#endif
#endif
#ifndef HAVE_TDM
#define HAVE_TDM 0
#endif

union frag_u { v16bf f; v8bf h[2]; };

// A-matrix 16x32 bf16 fragment (per ISA table):
// lanes 0-15:  row = lane,    K in {0..7, 16..23}
// lanes 16-31: row = lane-16, K in {8..15, 24..31}
__device__ __forceinline__ v16bf ld_frag_a(const bf16* base, int row, int stride) {
  const int lane = threadIdx.x & 31;
  const int r    = row + (lane & 15);
  const int hi   = lane >> 4;
  const bf16* p  = base + r * stride + hi * 8;
  frag_u u;
  u.h[0] = *(const v8bf*)(p);
  u.h[1] = *(const v8bf*)(p + 16);
  return u.f;
}

// B-matrix 32x16 bf16 fragment, stored row-major-by-N (Bt[n][k]):
// lanes 0-15: col=lane, K in {0..7, 16..23} ; lanes 16-31: col=lane-16, K in {8..15, 24..31}
__device__ __forceinline__ v16bf ld_frag_b(const bf16* base, int row, int stride) {
  const int lane = threadIdx.x & 31;
  const int r    = row + (lane & 15);
  const int hi   = lane >> 4;
  const bf16* p  = base + r * stride + hi * 8;
  frag_u u;
  u.h[0] = *(const v8bf*)(p);
  u.h[1] = *(const v8bf*)(p + 16);
  return u.f;
}

#if HAVE_TDM
// ---------------------------------------------------------------------------
// Tensor Data Mover: 2D tile (tile_x elems of 2B, tile_y rows) global -> LDS.
// LDS gets pad_amount(=4 DW =16B) inserted every pad_interval(=16 DW =64B),
// matching an LDS row stride of (tile_x + 8) bf16 when tile_x == 32.
// Descriptor packing per CDNA5 ISA ch.8 (D# group0 = 4 SGPRs, group1 = 8).
// ---------------------------------------------------------------------------
__device__ __forceinline__ void tdm_load_2d(const bf16* gptr, u32 lds_byte_off,
                                            u32 tile_x, u32 tile_y,
                                            u64 row_stride_elems) {
  const u64 ga = (u64)(uintptr_t)gptr;
  v4u g0;
  g0[0] = 1u;                                            // count=1, user-mode
  g0[1] = lds_byte_off;                                  // lds_addr
  g0[2] = (u32)(ga & 0xFFFFFFFFu);                       // global_addr[31:0]
  g0[3] = (u32)((ga >> 32) & 0x01FFFFFFu) | (2u << 30);  // addr[56:32] | type=2

  const u32 td0 = 0x7FFFFFFFu;                           // huge tensor dims:
  const u32 td1 = 0x7FFFFFFFu;                           // no OOB clipping
  v8i g1;
  g1[0] = (int)((1u << 16)        // data_size = 1 -> 2 bytes
              | (1u << 20)        // pad_enable
              | (3u << 22)        // pad_interval: 16 DWORDs (64B)
              | (3u << 25));      // pad_amount:    4 DWORDs (16B)
  g1[1] = (int)((td0 & 0xFFFFu) << 16);                  // tensor_dim0 lo
  g1[2] = (int)((td0 >> 16) | ((td1 & 0xFFFFu) << 16));  // dim0 hi | dim1 lo
  g1[3] = (int)((td1 >> 16) | (tile_x << 16));           // dim1 hi | tile_dim0
  g1[4] = (int)(tile_y & 0xFFFFu);                       // tile_dim1, tile_dim2=0
  const u64 st = row_stride_elems;
  g1[5] = (int)(u32)(st & 0xFFFFFFFFu);                  // dim0_stride[31:0]
  g1[6] = (int)(u32)((st >> 32) & 0xFFFFu);              // dim0_stride[47:32]
  g1[7] = 0;

  const v4i gz = {0, 0, 0, 0};
#if defined(__clang_major__) && (__clang_major__ >= 23)
  const v8i gz8 = {0, 0, 0, 0, 0, 0, 0, 0};
  __builtin_amdgcn_tensor_load_to_lds(g0, g1, gz, gz, gz8, 0);
#else
  __builtin_amdgcn_tensor_load_to_lds(g0, g1, gz, gz, 0);
#endif
}
#endif  // HAVE_TDM

// ---------------------------------------------------------------------------
// Generic NT GEMM: C[m,n] = epilogue( alpha * sum_k A[m,k]*B[n,k] + bias[n] )
// TDM double-buffered tile pipeline when available.
// ---------------------------------------------------------------------------
template<int WAVES_M, int WAVES_N, int WM, int WN>
__global__ __launch_bounds__(256) void gemm_nt_kernel(
    const bf16* __restrict__ A, int lda,
    const bf16* __restrict__ Bm, int ldb,
    int K, int ldc,
    float alpha, const float* __restrict__ bias, int relu,
    float* __restrict__ outf, bf16* __restrict__ outb,
    bf16* __restrict__ out_shift, int Lseq,
    const bf16* __restrict__ add_mat, const float* __restrict__ row_scale)
{
  constexpr int BM = 16 * WM * WAVES_M;
  constexpr int BN = 16 * WN * WAVES_N;
  constexpr int KB = 32;
  constexpr int LSTR = KB + 8;            // 40 bf16 = 64B data + 16B pad

  __shared__ __align__(16) bf16 sA[2][BM * LSTR];
  __shared__ __align__(16) bf16 sB[2][BN * LSTR];

  const int tid  = threadIdx.x;
  const int lane = tid & 31;
  const int wave = tid >> 5;
  const int wm   = wave / WAVES_N;
  const int wn   = wave % WAVES_N;
  const int rowBase = blockIdx.y * BM;
  const int colBase = blockIdx.x * BN;

  v8f acc[WM][WN];
#pragma unroll
  for (int m = 0; m < WM; ++m)
#pragma unroll
    for (int n = 0; n < WN; ++n) acc[m][n] = v8f{};

#if HAVE_TDM
  // prologue: DMA first K-slab into buffer 0 (wave 0 drives the TDM)
  if (wave == 0) {
    tdm_load_2d(A  + (size_t)rowBase * lda, (u32)(uintptr_t)&sA[0][0], KB, BM, (u64)lda);
    tdm_load_2d(Bm + (size_t)colBase * ldb, (u32)(uintptr_t)&sB[0][0], KB, BN, (u64)ldb);
  }
  int cur = 0;
  for (int k0 = 0; k0 < K; k0 += KB) {
    const int nxt = cur ^ 1;
    if (wave == 0) {
      if (k0 + KB < K) {
        tdm_load_2d(A  + (size_t)rowBase * lda + k0 + KB, (u32)(uintptr_t)&sA[nxt][0], KB, BM, (u64)lda);
        tdm_load_2d(Bm + (size_t)colBase * ldb + k0 + KB, (u32)(uintptr_t)&sB[nxt][0], KB, BN, (u64)ldb);
        __builtin_amdgcn_s_wait_tensorcnt(2);   // in-order: cur slab complete
      } else {
        __builtin_amdgcn_s_wait_tensorcnt(0);
      }
    }
    __syncthreads();            // publish cur tiles to all waves

    v16bf af[WM], bfr[WN];
#pragma unroll
    for (int m = 0; m < WM; ++m) af[m] = ld_frag_a(sA[cur], (wm * WM + m) * 16, LSTR);
#pragma unroll
    for (int n = 0; n < WN; ++n) bfr[n] = ld_frag_b(sB[cur], (wn * WN + n) * 16, LSTR);
#pragma unroll
    for (int m = 0; m < WM; ++m)
#pragma unroll
      for (int n = 0; n < WN; ++n)
        acc[m][n] = __builtin_amdgcn_wmma_f32_16x16x32_bf16(
            false, af[m], false, bfr[n], (short)0, acc[m][n], false, false);
    __syncthreads();            // everyone done reading cur before reuse
    cur = nxt;
  }
#else
  for (int k0 = 0; k0 < K; k0 += KB) {
#pragma unroll 2
    for (int i = tid; i < BM * (KB / 8); i += 256) {
      const int r = i >> 2;
      const int c = (i & 3) * 8;
      uint4 val = *(const uint4*)(A + (size_t)(rowBase + r) * lda + k0 + c);
      *(uint4*)(&sA[0][0] + r * LSTR + c) = val;
    }
#pragma unroll 2
    for (int i = tid; i < BN * (KB / 8); i += 256) {
      const int r = i >> 2;
      const int c = (i & 3) * 8;
      uint4 val = *(const uint4*)(Bm + (size_t)(colBase + r) * ldb + k0 + c);
      *(uint4*)(&sB[0][0] + r * LSTR + c) = val;
    }
    __syncthreads();
    v16bf af[WM], bfr[WN];
#pragma unroll
    for (int m = 0; m < WM; ++m) af[m] = ld_frag_a(&sA[0][0], (wm * WM + m) * 16, LSTR);
#pragma unroll
    for (int n = 0; n < WN; ++n) bfr[n] = ld_frag_b(&sB[0][0], (wn * WN + n) * 16, LSTR);
#pragma unroll
    for (int m = 0; m < WM; ++m)
#pragma unroll
      for (int n = 0; n < WN; ++n)
        acc[m][n] = __builtin_amdgcn_wmma_f32_16x16x32_bf16(
            false, af[m], false, bfr[n], (short)0, acc[m][n], false, false);
    __syncthreads();
  }
#endif

  // epilogue: C layout — VGPR i: lanes 0-15 row=i, lanes 16-31 row=i+8
  const int ln = lane & 15;
  const int lh = lane >> 4;
#pragma unroll
  for (int m = 0; m < WM; ++m) {
#pragma unroll
    for (int n = 0; n < WN; ++n) {
      const int col = colBase + (wn * WN + n) * 16 + ln;
      const float bv = bias ? bias[col] : 0.0f;
#pragma unroll
      for (int i = 0; i < 8; ++i) {
        const int row = rowBase + (wm * WM + m) * 16 + lh * 8 + i;
        float v = acc[m][n][i] * alpha + bv;
        if (relu) v = fmaxf(v, 0.0f);
        const size_t idx = (size_t)row * ldc + col;
        if (add_mat) v += row_scale[row] * (float)add_mat[idx];
        if (outf) outf[idx] = v;
        if (outb) outb[idx] = (bf16)v;
        if (out_shift) {
          if ((row % Lseq) != Lseq - 1) out_shift[idx + ldc] = (bf16)v;
        }
      }
    }
  }
}

// ---------------------------------------------------------------------------
// Flash-style causal attention. One block = (b, h, 16-row q tile).
// dh = 8 (scores scalar), attn@V (dv=512) via WMMA; V staged transposed.
// ---------------------------------------------------------------------------
__global__ __launch_bounds__(256) void attn_kernel(
    const float* __restrict__ qbuf,   // (MROWS, 64) f32
    const float* __restrict__ kbuf,   // (MROWS, 64) f32
    const bf16*  __restrict__ vbuf,   // (MROWS, WIDTH) bf16
    bf16* __restrict__ obuf)          // (MROWS, WIDTH) bf16
{
  __shared__ __align__(16) float sQ[16][DH];
  __shared__ __align__(16) float sK[32][DH];
  __shared__ float sS[16][32];
  __shared__ __align__(16) bf16 sP[16][48];
  __shared__ __align__(16) bf16 sVt[DV][40];
  __shared__ float sM[16], sL[16], sScale[16];

  const int tid  = threadIdx.x;
  const int lane = tid & 31;
  const int wave = tid >> 5;
  const int q0   = blockIdx.x * 16;
  const int h    = blockIdx.y;
  const int b    = blockIdx.z;
  const size_t rowB = (size_t)b * SEQL;
  const float scale = 0.35355339059327373f;   // 1/sqrt(8)

  if (tid < 16 * DH) {
    const int r = tid / DH, c = tid % DH;
    sQ[r][c] = qbuf[(rowB + q0 + r) * NE + h * DH + c];
  }
  if (tid < 16) { sM[tid] = -1e30f; sL[tid] = 0.0f; }

  v8f acc[4];
#pragma unroll
  for (int t = 0; t < 4; ++t) acc[t] = v8f{};
  __syncthreads();

  for (int s0 = 0; s0 <= q0 + 15; s0 += 32) {
    if (tid < 32 * DH) {
      const int s = tid / DH, c = tid % DH;
      sK[s][c] = kbuf[(rowB + s0 + s) * NE + h * DH + c];
    }
#pragma unroll 2
    for (int i = tid; i < 32 * 64; i += 256) {
      const int s  = i >> 6;
      const int cc = (i & 63) * 8;
      v8bf val = *(const v8bf*)(vbuf + (rowB + s0 + s) * WIDTH + h * DV + cc);
#pragma unroll
      for (int j = 0; j < 8; ++j) sVt[cc + j][s] = val[j];
    }
    __syncthreads();

#pragma unroll 2
    for (int i = tid; i < 512; i += 256) {
      const int r = i >> 5, c = i & 31;
      const int s = s0 + c;
      float v;
      if (s > q0 + r) {
        v = -1e30f;
      } else {
        float d = 0.0f;
#pragma unroll
        for (int e = 0; e < DH; ++e) d += sQ[r][e] * sK[c][e];
        v = d * scale;
      }
      sS[r][c] = v;
    }
    __syncthreads();

    if (tid < 16) {
      float mo = sM[tid];
      float mb = mo;
      for (int c = 0; c < 32; ++c) mb = fmaxf(mb, sS[tid][c]);
      const float sc = __expf(mo - mb);
      float ls = sL[tid] * sc;
      for (int c = 0; c < 32; ++c) {
        const float p = __expf(sS[tid][c] - mb);
        sP[tid][c] = (bf16)p;
        ls += p;
      }
      sM[tid] = mb; sL[tid] = ls; sScale[tid] = sc;
    }
    __syncthreads();

    const int lh = lane >> 4;
#pragma unroll
    for (int i = 0; i < 8; ++i) {
      const float fs = sScale[lh * 8 + i];
#pragma unroll
      for (int t = 0; t < 4; ++t) acc[t][i] *= fs;
    }
    v16bf af = ld_frag_a(&sP[0][0], 0, 48);
#pragma unroll
    for (int t = 0; t < 4; ++t) {
      v16bf bfr = ld_frag_b(&sVt[0][0], wave * 64 + t * 16, 40);
      acc[t] = __builtin_amdgcn_wmma_f32_16x16x32_bf16(
          false, af, false, bfr, (short)0, acc[t], false, false);
    }
    __syncthreads();
  }

  const int ln = lane & 15;
  const int lh = lane >> 4;
#pragma unroll
  for (int t = 0; t < 4; ++t) {
#pragma unroll
    for (int i = 0; i < 8; ++i) {
      const int rl  = lh * 8 + i;
      const float inv = 1.0f / sL[rl];
      const int col = h * DV + wave * 64 + t * 16 + ln;
      obuf[(rowB + q0 + rl) * WIDTH + col] = (bf16)(acc[t][i] * inv);
    }
  }
}

// ---------------------------------------------------------------------------
// Small helper kernels
// ---------------------------------------------------------------------------
__global__ void cvt_bf16_kernel(const float* __restrict__ in, bf16* __restrict__ out, int n) {
  const int i = blockIdx.x * 256 + threadIdx.x;
  if (i < n) out[i] = (bf16)in[i];
}

__global__ void cvt_bf16_t_kernel(const float* __restrict__ in, bf16* __restrict__ out,
                                  int R, int C) {
  const int i = blockIdx.x * 256 + threadIdx.x;
  if (i < R * C) {
    const int r = i / C, c = i % C;
    out[(size_t)c * R + r] = (bf16)in[i];
  }
}

__global__ void prep_x_kernel(const float* __restrict__ x, const float* __restrict__ bvec,
                              bf16* __restrict__ xc, int n) {
  const int i = blockIdx.x * 256 + threadIdx.x;
  if (i < n) xc[i] = (bf16)(x[i] - bvec[i & (DIMIN - 1)]);
}

__global__ void zero_ctx0_kernel(bf16* __restrict__ zctx) {
  const int i = blockIdx.x * 256 + threadIdx.x;
  if (i < BATCH * WIDTH) {
    const int bb = i / WIDTH, c = i % WIDTH;
    zctx[(size_t)bb * SEQL * WIDTH + c] = (bf16)0.0f;
  }
}

__global__ __launch_bounds__(256) void rowstats_kernel(
    const float* __restrict__ Dz, const float* __restrict__ x_input,
    const float* __restrict__ bvec, float* __restrict__ ps, bf16* __restrict__ x2)
{
  const int wave = threadIdx.x >> 5;
  const int lane = threadIdx.x & 31;
  const int row  = blockIdx.x * 8 + wave;
  const float* dz = Dz + (size_t)row * DIMIN;
  const float* xi = x_input + (size_t)row * DIMIN;
  float s2 = 0.0f, sd = 0.0f;
  for (int c = lane; c < DIMIN; c += 32) {
    const float d  = dz[c];
    const float xv = xi[c] - bvec[c];
    s2 += d * d;
    sd += d * xv;
  }
#pragma unroll
  for (int off = 16; off >= 1; off >>= 1) {
    s2 += __shfl_xor(s2, off, 32);
    sd += __shfl_xor(sd, off, 32);
  }
  const float nrm = sqrtf(s2) + 1e-6f;
  const float p   = sd / (nrm * nrm);
  if (lane == 0) ps[row] = p;
  for (int c = lane; c < DIMIN; c += 32) {
    const float xv = xi[c] - bvec[c];
    x2[(size_t)row * DIMIN + c] = (bf16)(xv - p * dz[c]);
  }
}

// ---------------------------------------------------------------------------
// Host launcher
// ---------------------------------------------------------------------------
static inline size_t align256(size_t x) { return (x + 255) & ~(size_t)255; }

extern "C" void kernel_launch(void* const* d_in, const int* in_sizes, int n_in,
                              void* d_out, int out_size, void* d_ws, size_t ws_size,
                              hipStream_t stream) {
  (void)in_sizes; (void)n_in; (void)out_size; (void)ws_size;

  const float* x_input = (const float*)d_in[0];
  const float* D       = (const float*)d_in[1];
  const float* bvec    = (const float*)d_in[2];
  const float* k_w     = (const float*)d_in[3];
  const float* k_b     = (const float*)d_in[4];
  const float* q_w     = (const float*)d_in[5];
  const float* q_b     = (const float*)d_in[6];
  const float* v_w     = (const float*)d_in[7];
  const float* v_b     = (const float*)d_in[8];
  const float* o_w     = (const float*)d_in[9];
  const float* o_b     = (const float*)d_in[10];
  float* out           = (float*)d_out;

  char* w = (char*)d_ws;
  auto take = [&](size_t bytes) { void* p = (void*)w; w += align256(bytes); return p; };
  bf16* Dbf   = (bf16*)take((size_t)WIDTH * DIMIN * 2);
  bf16* Dtbf  = (bf16*)take((size_t)DIMIN * WIDTH * 2);
  bf16* kwbf  = (bf16*)take((size_t)NE * WIDTH * 2);
  bf16* qwbf  = (bf16*)take((size_t)NE * WIDTH * 2);
  bf16* vwbf  = (bf16*)take((size_t)WIDTH * WIDTH * 2);
  bf16* owbf  = (bf16*)take((size_t)WIDTH * WIDTH * 2);
  bf16* xc    = (bf16*)take((size_t)MROWS * DIMIN * 2);
  bf16* zin   = (bf16*)take((size_t)MROWS * WIDTH * 2);
  bf16* zctx  = (bf16*)take((size_t)MROWS * WIDTH * 2);
  bf16* vb    = (bf16*)take((size_t)MROWS * WIDTH * 2);
  bf16* ob    = (bf16*)take((size_t)MROWS * WIDTH * 2);
  bf16* zpred = (bf16*)take((size_t)MROWS * WIDTH * 2);
  bf16* zsum  = (bf16*)take((size_t)MROWS * WIDTH * 2);
  bf16* x2    = (bf16*)take((size_t)MROWS * DIMIN * 2);
  float* kf   = (float*)take((size_t)MROWS * NE * 4);
  float* qf   = (float*)take((size_t)MROWS * NE * 4);
  float* Dzf  = (float*)take((size_t)MROWS * DIMIN * 4);
  float* psf  = (float*)take((size_t)MROWS * 4);

  const float lam = 1.0f / (4.0f * DIMIN);

  cvt_bf16_kernel<<<(WIDTH * DIMIN + 255) / 256, 256, 0, stream>>>(D, Dbf, WIDTH * DIMIN);
  cvt_bf16_t_kernel<<<(WIDTH * DIMIN + 255) / 256, 256, 0, stream>>>(D, Dtbf, WIDTH, DIMIN);
  cvt_bf16_kernel<<<(NE * WIDTH + 255) / 256, 256, 0, stream>>>(k_w, kwbf, NE * WIDTH);
  cvt_bf16_kernel<<<(NE * WIDTH + 255) / 256, 256, 0, stream>>>(q_w, qwbf, NE * WIDTH);
  cvt_bf16_kernel<<<(WIDTH * WIDTH + 255) / 256, 256, 0, stream>>>(v_w, vwbf, WIDTH * WIDTH);
  cvt_bf16_kernel<<<(WIDTH * WIDTH + 255) / 256, 256, 0, stream>>>(o_w, owbf, WIDTH * WIDTH);

  prep_x_kernel<<<(MROWS * DIMIN + 255) / 256, 256, 0, stream>>>(x_input, bvec, xc, MROWS * DIMIN);
  zero_ctx0_kernel<<<(BATCH * WIDTH + 255) / 256, 256, 0, stream>>>(zctx);

  #define GEMM_A gemm_nt_kernel<4, 2, 2, 4>
  #define GEMM_B gemm_nt_kernel<8, 1, 1, 4>

  // 1) z_input = relu(lam * xc @ D^T)   (+ shifted store into z_ctx)
  GEMM_A<<<dim3(WIDTH / 128, MROWS / 128), 256, 0, stream>>>(
      xc, DIMIN, Dbf, DIMIN, DIMIN, WIDTH,
      lam, nullptr, 1, nullptr, zin, zctx, SEQL, nullptr, nullptr);

  // 2) k = z_ctx @ k_w^T + k_b
  GEMM_B<<<dim3(NE / 64, MROWS / 128), 256, 0, stream>>>(
      zctx, WIDTH, kwbf, WIDTH, WIDTH, NE,
      1.0f, k_b, 0, kf, nullptr, nullptr, SEQL, nullptr, nullptr);

  // 3) q = z_input @ q_w^T + q_b
  GEMM_B<<<dim3(NE / 64, MROWS / 128), 256, 0, stream>>>(
      zin, WIDTH, qwbf, WIDTH, WIDTH, NE,
      1.0f, q_b, 0, qf, nullptr, nullptr, SEQL, nullptr, nullptr);

  // 4) v = z_ctx @ v_w^T + v_b
  GEMM_A<<<dim3(WIDTH / 128, MROWS / 128), 256, 0, stream>>>(
      zctx, WIDTH, vwbf, WIDTH, WIDTH, WIDTH,
      1.0f, v_b, 0, nullptr, vb, nullptr, SEQL, nullptr, nullptr);

  // 5) causal attention -> o
  attn_kernel<<<dim3(SEQL / 16, NHEADS, BATCH), 256, 0, stream>>>(qf, kf, vb, ob);

  // 6) z_pred_ = relu(o @ o_w^T + o_b)
  GEMM_A<<<dim3(WIDTH / 128, MROWS / 128), 256, 0, stream>>>(
      ob, WIDTH, owbf, WIDTH, WIDTH, WIDTH,
      1.0f, o_b, 1, nullptr, zpred, nullptr, SEQL, nullptr, nullptr);

  // 7) Dz = z_pred_ @ D   (NT against D^T copy)
  GEMM_A<<<dim3(DIMIN / 128, MROWS / 128), 256, 0, stream>>>(
      zpred, WIDTH, Dtbf, WIDTH, WIDTH, DIMIN,
      1.0f, nullptr, 0, Dzf, nullptr, nullptr, SEQL, nullptr, nullptr);

  // 8) proj_scale + x2 = x - ps*Dz
  rowstats_kernel<<<MROWS / 8, 256, 0, stream>>>(Dzf, x_input, bvec, psf, x2);

  // 9) zsum = relu(lam * x2 @ D^T) + ps * z_pred_
  GEMM_A<<<dim3(WIDTH / 128, MROWS / 128), 256, 0, stream>>>(
      x2, DIMIN, Dbf, DIMIN, DIMIN, WIDTH,
      lam, nullptr, 1, nullptr, zsum, nullptr, SEQL, zpred, psf);

  // 10) x_recons = zsum @ D + b   -> d_out
  GEMM_A<<<dim3(DIMIN / 128, MROWS / 128), 256, 0, stream>>>(
      zsum, WIDTH, Dtbf, WIDTH, WIDTH, DIMIN,
      1.0f, bvec, 0, out, nullptr, nullptr, SEQL, nullptr, nullptr);

  #undef GEMM_A
  #undef GEMM_B
}
